// MyDecoderWithAttention_78520592105859
// MI455X (gfx1250) — matrix operand order repaired and checked by reading, not compiled, not run on hardware
//
#include <hip/hip_runtime.h>
#include <hip/hip_bf16.h>
#include <math.h>
#include <stdint.h>

typedef __attribute__((ext_vector_type(16))) __bf16 v16bf;
typedef __attribute__((ext_vector_type(8)))  float  v8f;

#define UNITS  1024
#define ENC_D  1024
#define EMBED  256
#define VOCAB  50257
#define B_SZ   64
#define T_ENC  128
#define GRU_IN 1280
#define GRU3   3072

// ---- CDNA5 async global->LDS copy (ASYNCcnt path, ISA ch.08 / 15.18) ------
// VDST = LDS byte offset (low 32 bits of generic shared pointer), GV mode.
static __device__ inline void async_copy_b128(unsigned lds_off, const void* gptr) {
  asm volatile("global_load_async_to_lds_b128 %0, %1, off"
               :: "v"(lds_off), "v"(gptr) : "memory");
}
static __device__ inline void wait_async0() {
  asm volatile("s_wait_asynccnt 0x0" ::: "memory");
}

// ---- WMMA operand loaders -------------------------------------------------
// A fragment: 16x32 bf16 from row-major bf16. ISA layout (16-bit A 16x32):
// lanes 0-15: M=lane, VGPR0..3 = K 0..7, VGPR4..7 = K 16..23
// lanes 16-31: M=lane-16, VGPR0..3 = K 8..15, VGPR4..7 = K 24..31
static __device__ inline v16bf load_A16x32_bf(const __bf16* __restrict__ A, int lda,
                                              int row0, int kbase) {
  const int l = threadIdx.x & 31;
  const __bf16* base = A + (size_t)(row0 + (l & 15)) * lda + kbase + ((l >> 4) << 3);
  union { uint4 u[2]; v16bf v; } t;
  t.u[0] = *reinterpret_cast<const uint4*>(base);       // K+0 .. K+7
  t.u[1] = *reinterpret_cast<const uint4*>(base + 16);  // K+16 .. K+23
  return t.v;
}

// B fragment: 32x16 bf16 from an LDS slab stored transposed: ldsB[n][k].
// ISA layout (16-bit B 32x16): lanes 0-15 hold K=0..7,16..23 (N=lane),
// lanes 16-31 hold K=8..15,24..31 (N=lane-16), 2 values per VGPR in K order.
static __device__ inline v16bf load_B_lds(const __bf16* __restrict__ ldsB, int kpad,
                                          int kbase) {
  const int l = threadIdx.x & 31;
  const __bf16* p = ldsB + (size_t)(l & 15) * kpad + kbase + ((l >> 4) << 3);
  union { uint4 u[2]; v16bf v; } t;
  t.u[0] = *reinterpret_cast<const uint4*>(p);
  t.u[1] = *reinterpret_cast<const uint4*>(p + 16);
  return t.v;
}

// ---- Prep: LDS-tiled transpose+convert  out[N][K] bf16 = in[K][N] f32 -----
// grid = (N/32, K/32), block = 256. K, N multiples of 32.
__global__ void transpose_cvt_bf16(const float* __restrict__ in, __bf16* __restrict__ out,
                                   int K, int N) {
  __shared__ float tile[32][33];
  const int n0 = blockIdx.x * 32, k0 = blockIdx.y * 32;
  const int tx = threadIdx.x & 31, ty = threadIdx.x >> 5;
  for (int r = ty; r < 32; r += 8)
    tile[r][tx] = in[(size_t)(k0 + r) * N + n0 + tx];
  __syncthreads();
  for (int r = ty; r < 32; r += 8)
    out[(size_t)(n0 + r) * K + k0 + tx] = (__bf16)tile[tx][r];
}

// ---- Prep: elementwise f32 -> bf16 ---------------------------------------
__global__ void cvt_bf16_kernel(const float* __restrict__ in, __bf16* __restrict__ out,
                                int n) {
  const int i = blockIdx.x * blockDim.x + threadIdx.x;
  if (i < n) out[i] = (__bf16)in[i];
}

// ---- GEMM (async B): C(64xN) = A(64xK,bf16) @ Bt(NxK,bf16 pre-transposed) -
// grid.x = N/16, block = 128 (4 waves; wave w owns rows 16w..16w+15)
__global__ void wmma_gemm_async(const __bf16* __restrict__ A, const __bf16* __restrict__ Bt,
                                float* __restrict__ C, int N, int K,
                                const float* __restrict__ bias1,
                                const float* __restrict__ bias2) {
  extern __shared__ __bf16 ldsB[];
  const int kpad = K + 8;
  const unsigned lbase = (unsigned)(uintptr_t)ldsB;
  const int n0 = blockIdx.x * 16;
  const int tid = threadIdx.x;
  // Async-copy 16 rows of Bt (each K*2 bytes) into the padded LDS slab.
  const int gpr = (K * 2) >> 4;  // 16B granules per row
  for (int g = tid; g < 16 * gpr; g += blockDim.x) {
    const int n = g / gpr, kk = g - n * gpr;
    async_copy_b128(lbase + (unsigned)(n * kpad * 2 + kk * 16),
                    (const char*)Bt + (size_t)(n0 + n) * K * 2 + (size_t)kk * 16);
  }
  wait_async0();
  __syncthreads();
  const int w = tid >> 5;
  const int l = tid & 31;
  v8f acc = {0.f, 0.f, 0.f, 0.f, 0.f, 0.f, 0.f, 0.f};
  for (int kb = 0; kb < K; kb += 32) {
    v16bf a  = load_A16x32_bf(A, K, w * 16, kb);
    v16bf bb = load_B_lds(ldsB, kpad, kb);
    acc = __builtin_amdgcn_wmma_f32_16x16x32_bf16(false, a, false, bb,
                                                  (short)0, acc, false, false);
  }
  const int col = n0 + (l & 15);
  const float bias = (bias1 ? bias1[col] : 0.0f) + (bias2 ? bias2[col] : 0.0f);
  const int rbase = w * 16 + ((l >> 4) << 3);
#pragma unroll
  for (int v = 0; v < 8; v++)
    C[(size_t)(rbase + v) * N + col] = acc[v] + bias;
}

// ---- GEMM (cvt-staged B): for single-use f32 B (fcW). Tail-guarded N. -----
__global__ void wmma_gemm_bias(const __bf16* __restrict__ A, const float* __restrict__ Bm,
                               float* __restrict__ C, int N, int K,
                               const float* __restrict__ bias1) {
  extern __shared__ __bf16 ldsB[];
  const int kpad = K + 8;
  const int n0 = blockIdx.x * 16;
  const int tid = threadIdx.x;
  for (int idx = tid; idx < 16 * K; idx += blockDim.x) {
    const int n = idx & 15, k = idx >> 4;
    const int col = n0 + n;
    const float val = (col < N) ? Bm[(size_t)k * N + col] : 0.0f;
    ldsB[(size_t)n * kpad + k] = (__bf16)val;
  }
  __syncthreads();
  const int w = tid >> 5;
  const int l = tid & 31;
  v8f acc = {0.f, 0.f, 0.f, 0.f, 0.f, 0.f, 0.f, 0.f};
  for (int kb = 0; kb < K; kb += 32) {
    v16bf a  = load_A16x32_bf(A, K, w * 16, kb);
    v16bf bb = load_B_lds(ldsB, kpad, kb);
    acc = __builtin_amdgcn_wmma_f32_16x16x32_bf16(false, a, false, bb,
                                                  (short)0, acc, false, false);
  }
  const int col = n0 + (l & 15);
  if (col < N) {
    const float bias = bias1 ? bias1[col] : 0.0f;
    const int rbase = w * 16 + ((l >> 4) << 3);
#pragma unroll
    for (int v = 0; v < 8; v++)
      C[(size_t)(rbase + v) * N + col] = acc[v] + bias;
  }
}

// ---- Fused attention score: score[b,t] = sum_u tanh(enc@W2 + q)[u]*V[u] ---
// grid.x = B, block = 256 (8 waves; wave w owns t rows 16w..16w+15).
// q already contains hidden@W1 + b1 + b2; k (8192x1024) never materialized.
__global__ void score_kernel(const __bf16* __restrict__ enc_bf,
                             const __bf16* __restrict__ W2T,
                             const float* __restrict__ qplus, const float* __restrict__ V,
                             const float* __restrict__ bV, float* __restrict__ score) {
  extern __shared__ __bf16 ldsB[];
  const int kpad = UNITS + 8;
  const unsigned lbase = (unsigned)(uintptr_t)ldsB;
  const int b = blockIdx.x;
  const int tid = threadIdx.x;
  const int w = tid >> 5, l = tid & 31;
  const __bf16* encB = enc_bf + (size_t)b * T_ENC * ENC_D;
  float part[8] = {0.f, 0.f, 0.f, 0.f, 0.f, 0.f, 0.f, 0.f};
  for (int nt = 0; nt < UNITS / 16; nt++) {
    // Async-stage W2T rows nt*16..+16 (each 2KB) into LDS.
    for (int g = tid; g < 16 * 128; g += 256) {       // 128 granules per row
      const int n = g >> 7, kk = g & 127;
      async_copy_b128(lbase + (unsigned)(n * kpad * 2 + kk * 16),
                      (const char*)W2T + (size_t)(nt * 16 + n) * UNITS * 2 + (size_t)kk * 16);
    }
    wait_async0();
    __syncthreads();
    v8f acc = {0.f, 0.f, 0.f, 0.f, 0.f, 0.f, 0.f, 0.f};
    for (int kb = 0; kb < ENC_D; kb += 32) {
      v16bf a  = load_A16x32_bf(encB, ENC_D, w * 16, kb);
      v16bf bb = load_B_lds(ldsB, kpad, kb);
      acc = __builtin_amdgcn_wmma_f32_16x16x32_bf16(false, a, false, bb,
                                                    (short)0, acc, false, false);
    }
    const int u = nt * 16 + (l & 15);
    const float qv = qplus[(size_t)b * UNITS + u];
    const float vv = V[u];
#pragma unroll
    for (int v = 0; v < 8; v++) part[v] += tanhf(acc[v] + qv) * vv;
    __syncthreads();
  }
  // Reduce over the 16 lanes sharing each row (xor 1,2,4,8 keeps halves apart).
#pragma unroll
  for (int v = 0; v < 8; v++) {
    float p = part[v];
    p += __shfl_xor(p, 1, 32);
    p += __shfl_xor(p, 2, 32);
    p += __shfl_xor(p, 4, 32);
    p += __shfl_xor(p, 8, 32);
    part[v] = p;
  }
  if ((l & 15) == 0) {
    const int t = w * 16 + ((l >> 4) << 3);
#pragma unroll
    for (int v = 0; v < 8; v++)
      score[(size_t)b * T_ENC + t + v] = part[v] + bV[0];
  }
}

// ---- Softmax over T, context = attw^T . enc, embedding gather -> xin(bf16) -
__global__ void softmax_context_kernel(const float* __restrict__ score,
                                       const float* __restrict__ enc,
                                       const float* __restrict__ emb,
                                       const int* __restrict__ x,
                                       float* __restrict__ attw_out,
                                       __bf16* __restrict__ xin_bf) {
  __shared__ float sc[T_ENC];
  __shared__ float stat[2];
  const int b = blockIdx.x, tid = threadIdx.x;
  if (tid < T_ENC) sc[tid] = score[(size_t)b * T_ENC + tid];
  __syncthreads();
  if (tid == 0) {
    float m = sc[0];
    for (int t = 1; t < T_ENC; t++) m = fmaxf(m, sc[t]);
    stat[0] = m;
  }
  __syncthreads();
  if (tid < T_ENC) sc[tid] = expf(sc[tid] - stat[0]);
  __syncthreads();
  if (tid == 0) {
    float s = 0.f;
    for (int t = 0; t < T_ENC; t++) s += sc[t];
    stat[1] = s;
  }
  __syncthreads();
  if (tid < T_ENC) {
    const float a = sc[tid] / stat[1];
    sc[tid] = a;
    attw_out[(size_t)b * T_ENC + tid] = a;
  }
  __syncthreads();
  const float* encB = enc + (size_t)b * T_ENC * ENC_D;
  for (int d = tid; d < ENC_D; d += blockDim.x) {
    float acc = 0.f;
    for (int t = 0; t < T_ENC; t++) acc += sc[t] * encB[(size_t)t * ENC_D + d];
    xin_bf[(size_t)b * GRU_IN + d] = (__bf16)acc;
  }
  const int xi = x[b];
  for (int j = tid; j < EMBED; j += blockDim.x)
    xin_bf[(size_t)b * GRU_IN + ENC_D + j] = (__bf16)emb[(size_t)xi * EMBED + j];
}

// ---- GRU gates with h0 == 0 (recurrent matmul degenerates to gru_bias[1]) -
__global__ void gru_gate_kernel(const float* __restrict__ xzrh,
                                const float* __restrict__ gbias,
                                float* __restrict__ h_out,
                                __bf16* __restrict__ h_bf) {
  const int idx = blockIdx.x * blockDim.x + threadIdx.x;
  if (idx >= B_SZ * UNITS) return;
  const int b = idx / UNITS, u = idx % UNITS;
  const float* g0 = gbias;
  const float* g1 = gbias + GRU3;
  const float* xx = xzrh + (size_t)b * GRU3;
  const float z = 1.f / (1.f + expf(-(xx[u] + g0[u] + g1[u])));
  const float r = 1.f / (1.f + expf(-(xx[UNITS + u] + g0[UNITS + u] + g1[UNITS + u])));
  const float hh = tanhf(xx[2 * UNITS + u] + g0[2 * UNITS + u] + r * g1[2 * UNITS + u]);
  const float h = (1.f - z) * hh;  // h = z*h0 + (1-z)*hh, h0 = 0
  h_out[idx] = h;
  h_bf[idx] = (__bf16)h;
}

extern "C" void kernel_launch(void* const* d_in, const int* in_sizes, int n_in,
                              void* d_out, int out_size, void* d_ws, size_t ws_size,
                              hipStream_t stream) {
  (void)in_sizes; (void)n_in; (void)out_size; (void)ws_size;
  const int*   x      = (const int*)  d_in[0];
  const float* hidden = (const float*)d_in[1];
  const float* enc    = (const float*)d_in[2];
  const float* emb    = (const float*)d_in[3];
  const float* W1     = (const float*)d_in[4];
  const float* b1     = (const float*)d_in[5];
  const float* W2     = (const float*)d_in[6];
  const float* b2     = (const float*)d_in[7];
  const float* V      = (const float*)d_in[8];
  const float* bV     = (const float*)d_in[9];
  const float* gk     = (const float*)d_in[10];
  // d_in[11] = gru_rkernel: unused (h0 == 0 so 0 @ gru_rkernel == 0)
  const float* gbias  = (const float*)d_in[12];
  const float* fcW    = (const float*)d_in[13];
  const float* fcb    = (const float*)d_in[14];

  float* out    = (float*)d_out;
  float* logits = out;                              // 64 * 50257
  float* h_out  = out + (size_t)B_SZ * VOCAB;       // 64 * 1024
  float* attw   = h_out + (size_t)B_SZ * UNITS;     // 64 * 128

  // Workspace layout (f32 region first, then 16B-aligned bf16 region).
  float*  q      = (float*)d_ws;                    // 64*1024
  float*  score  = q + (size_t)B_SZ * UNITS;        // 64*128
  float*  xzrh   = score + (size_t)B_SZ * T_ENC;    // 64*3072
  __bf16* enc_bf = (__bf16*)(xzrh + (size_t)B_SZ * GRU3);
  __bf16* hid_bf = enc_bf + (size_t)B_SZ * T_ENC * ENC_D;
  __bf16* xin_bf = hid_bf + (size_t)B_SZ * UNITS;
  __bf16* h_bf   = xin_bf + (size_t)B_SZ * GRU_IN;
  __bf16* W1T    = h_bf + (size_t)B_SZ * UNITS;     // [N=1024][K=1024]
  __bf16* W2T    = W1T + (size_t)UNITS * UNITS;     // [N=1024][K=1024]
  __bf16* gkT    = W2T + (size_t)UNITS * UNITS;     // [N=3072][K=1280]

  const size_t sh1024 = (size_t)16 * (UNITS + 8) * sizeof(__bf16);
  const size_t sh1280 = (size_t)16 * (GRU_IN + 8) * sizeof(__bf16);

  // 0) one-time (per call) precision/layout prep
  transpose_cvt_bf16<<<dim3(UNITS / 32, UNITS / 32), 256, 0, stream>>>(W1, W1T, UNITS, UNITS);
  transpose_cvt_bf16<<<dim3(UNITS / 32, UNITS / 32), 256, 0, stream>>>(W2, W2T, UNITS, UNITS);
  transpose_cvt_bf16<<<dim3(GRU3 / 32, GRU_IN / 32), 256, 0, stream>>>(gk, gkT, GRU_IN, GRU3);
  cvt_bf16_kernel<<<(B_SZ * T_ENC * ENC_D) / 256, 256, 0, stream>>>(enc, enc_bf,
                                                                    B_SZ * T_ENC * ENC_D);
  cvt_bf16_kernel<<<(B_SZ * UNITS) / 256, 256, 0, stream>>>(hidden, hid_bf, B_SZ * UNITS);

  // 1) q = hidden @ W1 + b1 + b2 (b2 folded so score epilogue is q+k)
  wmma_gemm_async<<<UNITS / 16, 128, sh1024, stream>>>(hid_bf, W1T, q, UNITS, UNITS, b1, b2);
  // 2) fused k-GEMM + tanh + V-dot -> score (17.2 GF bf16 WMMA, async B staging)
  score_kernel<<<B_SZ, 256, sh1024, stream>>>(enc_bf, W2T, q, V, bV, score);
  // 3) softmax + context + embedding gather -> xin (bf16)
  softmax_context_kernel<<<B_SZ, 256, 0, stream>>>(score, enc, emb, x, attw, xin_bf);
  // 4) xin @ gru_kernel -> [xz|xr|xh]
  wmma_gemm_async<<<GRU3 / 16, 128, sh1280, stream>>>(xin_bf, gkT, xzrh, GRU3, GRU_IN,
                                                      nullptr, nullptr);
  // 5) gates -> h (f32 output + bf16 copy for logits GEMM)
  gru_gate_kernel<<<(B_SZ * UNITS + 255) / 256, 256, 0, stream>>>(xzrh, gbias, h_out, h_bf);
  // 6) logits = h @ fcW + fcb (memory-bound: streams 206 MB of fcW once)
  wmma_gemm_bias<<<(VOCAB + 15) / 16, 128, sh1024, stream>>>(h_bf, fcW, logits, VOCAB, UNITS, fcb);
}
